// MultiHeadAttention_30751965839397
// MI455X (gfx1250) — compile-verified, matrix operand order checked by reading, not run on hardware
//
#include <hip/hip_runtime.h>

// MI455X (gfx1250) multi-head attention, bf16 WMMA path.
// L=4096, D=1024, H=16, dk=64.
// Softmax scale 1/sqrt(dk)=0.125 is folded into the Q projection (exact,
// power of two), so the attention inner loop has no post-WMMA scale pass.

#define L_SEQ 4096
#define D_MODEL 1024
#define N_HEADS 16
#define D_K 64

typedef __attribute__((ext_vector_type(16))) __bf16 bf16x16;
typedef __attribute__((ext_vector_type(8)))  float  f32x8;

__device__ __forceinline__ unsigned short f32_to_bf16_rne(float f) {
    unsigned u = __builtin_bit_cast(unsigned, f);
    unsigned r = (u + 0x7FFFu + ((u >> 16) & 1u)) >> 16;
    return (unsigned short)r;
}

// ---------------------------------------------------------------------------
// f32 -> bf16 conversion, 4 elements / thread (float4 in, 2x packed u32 out)
// ---------------------------------------------------------------------------
__global__ __launch_bounds__(256)
void mha_cvt_bf16(const float4* __restrict__ src, uint2* __restrict__ dst, int n4) {
    int i = blockIdx.x * blockDim.x + threadIdx.x;
    if (i < n4) {
        float4 f = src[i];
        uint2 o;
        o.x = (unsigned)f32_to_bf16_rne(f.x) | ((unsigned)f32_to_bf16_rne(f.y) << 16);
        o.y = (unsigned)f32_to_bf16_rne(f.z) | ((unsigned)f32_to_bf16_rne(f.w) << 16);
        dst[i] = o;
    }
}

// ---------------------------------------------------------------------------
// GEMM: C[M,N] = (A[M,K](bf16) @ B[N,K](bf16)^T + bias) * oscale
//   MODE 0: f32 row-major [M,N]
//   MODE 1: bf16 head-split [H][L][dk]            (Q, K projections)
//   MODE 2: bf16 head-split transposed [H][dk][L] (V projection)
// Block tile 128x128, K-step 32, 256 threads = 8 waves (4m x 2n),
// wave tile 32x64 = 2x4 wmma frags. Double-buffered LDS; global loads for
// tile i+1 issue before the WMMAs of tile i, LDS stores after. All 12
// fragment ds_loads issue up-front so the 8 WMMAs stream with staged waits.
// ---------------------------------------------------------------------------
#define GBM 128
#define GBN 128
#define GBK 32
#define LDSTR 40                 // GBK + 8 pad (bf16 elems) vs bank conflicts
#define TILE_E (GBM * LDSTR)     // elems per LDS buffer

template <int MODE>
__global__ __launch_bounds__(256)
void mha_gemm_bf16(const unsigned short* __restrict__ A,
                   const unsigned short* __restrict__ B,
                   const float* __restrict__ bias,
                   float* __restrict__ outF,
                   unsigned short* __restrict__ outB,
                   int M, int N, int K, float oscale) {
    __shared__ __attribute__((aligned(16))) unsigned short As[2 * TILE_E];
    __shared__ __attribute__((aligned(16))) unsigned short Bs[2 * TILE_E];

    const int tid   = threadIdx.x;
    const int wave  = tid >> 5;
    const int lane  = tid & 31;
    const int wm    = wave >> 1;        // 0..3 -> 32-row band
    const int wn    = wave & 1;         // 0..1 -> 64-col band
    const int l16   = lane & 15;
    const int lhalf = lane >> 4;
    const int kbase = lhalf * 8;        // ISA 16-bit A/B lane K striping
    const int m_blk = blockIdx.x * GBM;
    const int n_blk = blockIdx.y * GBN;

    // staging map: thread -> one 32B row chunk of A and of B
    const int srow = tid >> 1;          // 0..127
    const int scol = (tid & 1) * 16;    // 0 or 16 (elems)

    f32x8 acc[2][4];
#pragma unroll
    for (int i = 0; i < 2; ++i)
#pragma unroll
        for (int j = 0; j < 4; ++j)
            acc[i][j] = f32x8{0.f,0.f,0.f,0.f,0.f,0.f,0.f,0.f};

    uint4 ra0, ra1, rb0, rb1;
    auto stage_load = [&](int k0) {
        const unsigned short* ap = A + (size_t)(m_blk + srow) * K + k0 + scol;
        const unsigned short* bp = B + (size_t)(n_blk + srow) * K + k0 + scol;
        ra0 = *(const uint4*)(ap);
        ra1 = *(const uint4*)(ap + 8);
        rb0 = *(const uint4*)(bp);
        rb1 = *(const uint4*)(bp + 8);
    };
    auto stage_store = [&](int buf) {
        unsigned short* as = &As[buf * TILE_E + srow * LDSTR + scol];
        unsigned short* bs = &Bs[buf * TILE_E + srow * LDSTR + scol];
        *(uint4*)(as)     = ra0;
        *(uint4*)(as + 8) = ra1;
        *(uint4*)(bs)     = rb0;
        *(uint4*)(bs + 8) = rb1;
    };

    stage_load(0);
    stage_store(0);
    __syncthreads();

    int cur = 0;
    for (int k0 = 0; k0 < K; k0 += GBK) {
        const bool has_next = (k0 + GBK) < K;
        if (has_next) stage_load(k0 + GBK);     // issue global loads early

        union Fr { bf16x16 v; uint4 u[2]; };
        Fr af[2], bfr[4];
        // All 12 fragment loads issue before any WMMA (distinct registers).
#pragma unroll
        for (int fm = 0; fm < 2; ++fm) {
            const unsigned short* as =
                &As[cur * TILE_E + (wm * 32 + fm * 16 + l16) * LDSTR + kbase];
            af[fm].u[0] = *(const uint4*)(as);
            af[fm].u[1] = *(const uint4*)(as + 16);
        }
#pragma unroll
        for (int fn = 0; fn < 4; ++fn) {
            const unsigned short* bs =
                &Bs[cur * TILE_E + (wn * 64 + fn * 16 + l16) * LDSTR + kbase];
            bfr[fn].u[0] = *(const uint4*)(bs);
            bfr[fn].u[1] = *(const uint4*)(bs + 16);
        }
#pragma unroll
        for (int fm = 0; fm < 2; ++fm)
#pragma unroll
            for (int fn = 0; fn < 4; ++fn)
                acc[fm][fn] = __builtin_amdgcn_wmma_f32_16x16x32_bf16(
                    false, af[fm].v, false, bfr[fn].v, (short)0, acc[fm][fn], false, false);

        if (has_next) {
            stage_store(cur ^ 1);               // waits loadcnt here, after WMMAs
            cur ^= 1;
        }
        __syncthreads();
    }

    // Epilogue: (acc + bias) * oscale, layout-specific store (MODE compile-time).
#pragma unroll
    for (int fm = 0; fm < 2; ++fm) {
#pragma unroll
        for (int fn = 0; fn < 4; ++fn) {
            int col = n_blk + wn * 64 + fn * 16 + l16;
            float bv = bias[col];
#pragma unroll
            for (int r = 0; r < 8; ++r) {
                int row = m_blk + wm * 32 + fm * 16 + r + lhalf * 8;
                float v = (acc[fm][fn][r] + bv) * oscale;
                if (MODE == 0) {
                    outF[(size_t)row * N + col] = v;
                } else if (MODE == 1) {        // [H][L][dk]
                    outB[(size_t)(col >> 6) * (L_SEQ * D_K) +
                         (size_t)row * D_K + (col & 63)] = f32_to_bf16_rne(v);
                } else {                        // [H][dk][L]
                    outB[(size_t)(col >> 6) * (D_K * L_SEQ) +
                         (size_t)(col & 63) * L_SEQ + row] = f32_to_bf16_rne(v);
                }
            }
        }
    }
}

// ---------------------------------------------------------------------------
// Flash-style attention: one head / 64 query rows per block.
// 128 threads = 4 waves; wave handles 16 query rows.
// K-frag loads batched before S-WMMAs; V-frag loads issued before softmax so
// their latency hides under the exp/shuffle VALU work. P goes C-layout ->
// A-layout through a per-wave LDS tile (DS ops are in-order per wave).
// Q already carries the 1/sqrt(dk) scale, so S is used directly.
// ---------------------------------------------------------------------------
__global__ __launch_bounds__(128)
void mha_attn(const unsigned short* __restrict__ Qh,  // [H][L][dk] (pre-scaled)
              const unsigned short* __restrict__ Kh,  // [H][L][dk]
              const unsigned short* __restrict__ Vt,  // [H][dk][L]
              unsigned short* __restrict__ Ao) {      // [L][D] bf16
    __shared__ __attribute__((aligned(16))) unsigned short Pbuf[4][16][64];

    const int tid   = threadIdx.x;
    const int wave  = tid >> 5;
    const int lane  = tid & 31;
    const int l16   = lane & 15;
    const int lhalf = lane >> 4;
    const int kbase = lhalf * 8;
    const int h     = blockIdx.y;
    const int q0    = blockIdx.x * 64 + wave * 16;

    const unsigned short* Qb = Qh + (size_t)h * L_SEQ * D_K;
    const unsigned short* Kb = Kh + (size_t)h * L_SEQ * D_K;
    const unsigned short* Vb = Vt + (size_t)h * D_K * L_SEQ;

    union F { bf16x16 v; uint4 u[2]; };

    // Q fragments (dk=64 -> two K=32 slabs), kept in registers.
    F aq[2];
#pragma unroll
    for (int kk = 0; kk < 2; ++kk) {
        const unsigned short* p = Qb + (size_t)(q0 + l16) * D_K + kk * 32 + kbase;
        aq[kk].u[0] = *(const uint4*)(p);
        aq[kk].u[1] = *(const uint4*)(p + 16);
    }

    f32x8 o[4];
#pragma unroll
    for (int n = 0; n < 4; ++n) o[n] = f32x8{0.f,0.f,0.f,0.f,0.f,0.f,0.f,0.f};
    float mrow[8], lrow[8];
#pragma unroll
    for (int r = 0; r < 8; ++r) { mrow[r] = -3.0e38f; lrow[r] = 0.f; }

    for (int k0 = 0; k0 < L_SEQ; k0 += 64) {
        if (k0 + 64 < L_SEQ) {  // gfx1250 global_prefetch for next key/value rows
            __builtin_prefetch(Kb + (size_t)(k0 + 64 + l16) * D_K, 0, 2);
            __builtin_prefetch(Vb + (size_t)l16 * L_SEQ + k0 + 64, 0, 2);
        }

        // ---- batched K fragment loads (16 x b128) ----------------------
        F bk[4][2];
#pragma unroll
        for (int n = 0; n < 4; ++n)
#pragma unroll
            for (int kk = 0; kk < 2; ++kk) {
                const unsigned short* p =
                    Kb + (size_t)(k0 + n * 16 + l16) * D_K + kk * 32 + kbase;
                bk[n][kk].u[0] = *(const uint4*)(p);
                bk[n][kk].u[1] = *(const uint4*)(p + 16);
            }

        // ---- S = Q K^T over a 16x64 key block (scale already in Q) -----
        f32x8 s[4];
#pragma unroll
        for (int n = 0; n < 4; ++n) {
            s[n] = f32x8{0.f,0.f,0.f,0.f,0.f,0.f,0.f,0.f};
#pragma unroll
            for (int kk = 0; kk < 2; ++kk)
                s[n] = __builtin_amdgcn_wmma_f32_16x16x32_bf16(
                    false, aq[kk].v, false, bk[n][kk].v, (short)0, s[n], false, false);
        }

        // ---- issue V fragment loads early (latency hides under softmax)
        F bv[4][2];
#pragma unroll
        for (int n = 0; n < 4; ++n)
#pragma unroll
            for (int kk = 0; kk < 2; ++kk) {
                const unsigned short* p =
                    Vb + (size_t)(n * 16 + l16) * L_SEQ + k0 + kk * 32 + kbase;
                bv[n][kk].u[0] = *(const uint4*)(p);
                bv[n][kk].u[1] = *(const uint4*)(p + 16);
            }

        // ---- Online softmax over key block -----------------------------
#pragma unroll
        for (int r = 0; r < 8; ++r) {
            float mx = fmaxf(fmaxf(s[0][r], s[1][r]), fmaxf(s[2][r], s[3][r]));
#pragma unroll
            for (int off = 8; off > 0; off >>= 1)
                mx = fmaxf(mx, __shfl_xor(mx, off, 16));
            float mnew  = fmaxf(mrow[r], mx);
            float alpha = __expf(mrow[r] - mnew);
            mrow[r] = mnew;
            float rs = 0.f;
#pragma unroll
            for (int n = 0; n < 4; ++n) {
                float p = __expf(s[n][r] - mnew);
                rs += p;
                Pbuf[wave][r + lhalf * 8][n * 16 + l16] = f32_to_bf16_rne(p);
            }
#pragma unroll
            for (int off = 8; off > 0; off >>= 1)
                rs += __shfl_xor(rs, off, 16);
            lrow[r] = lrow[r] * alpha + rs;
#pragma unroll
            for (int n = 0; n < 4; ++n) o[n][r] *= alpha;
        }

        // ---- O += P * V  (P re-read in A-layout from LDS) --------------
        F ap[2];
#pragma unroll
        for (int kk = 0; kk < 2; ++kk) {
            const unsigned short* p = &Pbuf[wave][l16][kk * 32 + kbase];
            ap[kk].u[0] = *(const uint4*)(p);
            ap[kk].u[1] = *(const uint4*)(p + 16);
        }
#pragma unroll
        for (int n = 0; n < 4; ++n)
#pragma unroll
            for (int kk = 0; kk < 2; ++kk)
                o[n] = __builtin_amdgcn_wmma_f32_16x16x32_bf16(
                    false, ap[kk].v, false, bv[n][kk].v, (short)0, o[n], false, false);
    }

    // ---- Finalize: divide by row sum, store bf16 [L][D] ----------------
#pragma unroll
    for (int r = 0; r < 8; ++r) {
        float inv = 1.0f / lrow[r];
        int row = q0 + r + lhalf * 8;
#pragma unroll
        for (int n = 0; n < 4; ++n) {
            float v = o[n][r] * inv;
            Ao[(size_t)row * D_MODEL + h * D_K + n * 16 + l16] = f32_to_bf16_rne(v);
        }
    }
}

// ---------------------------------------------------------------------------
extern "C" void kernel_launch(void* const* d_in, const int* in_sizes, int n_in,
                              void* d_out, int out_size, void* d_ws, size_t ws_size,
                              hipStream_t stream) {
    const float* q  = (const float*)d_in[0];
    const float* k  = (const float*)d_in[1];
    const float* v  = (const float*)d_in[2];
    const float* wq = (const float*)d_in[3];
    const float* bq = (const float*)d_in[4];
    const float* wk = (const float*)d_in[5];
    const float* bk = (const float*)d_in[6];
    const float* wv = (const float*)d_in[7];
    const float* bv = (const float*)d_in[8];
    const float* wo = (const float*)d_in[9];
    const float* bo = (const float*)d_in[10];
    float* out = (float*)d_out;

    const size_t LD = (size_t)L_SEQ * D_MODEL;   // 4M elems
    const size_t DD = (size_t)D_MODEL * D_MODEL; // 1M elems
    char* ws = (char*)d_ws;
    const size_t MB = 1u << 20;
    unsigned short* qbf  = (unsigned short*)(ws + 0 * MB);
    unsigned short* kbf  = (unsigned short*)(ws + 8 * MB);
    unsigned short* vbf  = (unsigned short*)(ws + 16 * MB);
    unsigned short* wqb  = (unsigned short*)(ws + 24 * MB);
    unsigned short* wkb  = (unsigned short*)(ws + 26 * MB);
    unsigned short* wvb  = (unsigned short*)(ws + 28 * MB);
    unsigned short* wob  = (unsigned short*)(ws + 30 * MB);
    unsigned short* Qh   = (unsigned short*)(ws + 32 * MB); // [H][L][dk]
    unsigned short* Kh   = (unsigned short*)(ws + 40 * MB); // [H][L][dk]
    unsigned short* Vt   = (unsigned short*)(ws + 48 * MB); // [H][dk][L]
    unsigned short* Aout = (unsigned short*)(ws + 56 * MB); // [L][D]

    // 1) f32 -> bf16 conversions (4 elems/thread)
    mha_cvt_bf16<<<(int)(LD / 4 / 256), 256, 0, stream>>>((const float4*)q,  (uint2*)qbf, (int)(LD / 4));
    mha_cvt_bf16<<<(int)(LD / 4 / 256), 256, 0, stream>>>((const float4*)k,  (uint2*)kbf, (int)(LD / 4));
    mha_cvt_bf16<<<(int)(LD / 4 / 256), 256, 0, stream>>>((const float4*)v,  (uint2*)vbf, (int)(LD / 4));
    mha_cvt_bf16<<<(int)(DD / 4 / 256), 256, 0, stream>>>((const float4*)wq, (uint2*)wqb, (int)(DD / 4));
    mha_cvt_bf16<<<(int)(DD / 4 / 256), 256, 0, stream>>>((const float4*)wk, (uint2*)wkb, (int)(DD / 4));
    mha_cvt_bf16<<<(int)(DD / 4 / 256), 256, 0, stream>>>((const float4*)wv, (uint2*)wvb, (int)(DD / 4));
    mha_cvt_bf16<<<(int)(DD / 4 / 256), 256, 0, stream>>>((const float4*)wo, (uint2*)wob, (int)(DD / 4));

    // 2) Projections; Q carries the 1/sqrt(dk)=0.125 softmax scale (exact pow2)
    dim3 ggrid(L_SEQ / GBM, D_MODEL / GBN);
    mha_gemm_bf16<1><<<ggrid, 256, 0, stream>>>(qbf, wqb, bq, nullptr, Qh,
                                                L_SEQ, D_MODEL, D_MODEL, 0.125f);
    mha_gemm_bf16<1><<<ggrid, 256, 0, stream>>>(kbf, wkb, bk, nullptr, Kh,
                                                L_SEQ, D_MODEL, D_MODEL, 1.0f);
    mha_gemm_bf16<2><<<ggrid, 256, 0, stream>>>(vbf, wvb, bv, nullptr, Vt,
                                                L_SEQ, D_MODEL, D_MODEL, 1.0f);

    // 3) Attention (flash-style, online softmax)
    mha_attn<<<dim3(L_SEQ / 64, N_HEADS), 128, 0, stream>>>(Qh, Kh, Vt, Aout);

    // 4) Output projection -> f32 d_out
    mha_gemm_bf16<0><<<ggrid, 256, 0, stream>>>(Aout, wob, bo, out, nullptr,
                                                L_SEQ, D_MODEL, D_MODEL, 1.0f);
}
